// MultiHeadAttention_33079838114174
// MI455X (gfx1250) — matrix operand order repaired and checked by reading, not compiled, not run on hardware
//
#include <hip/hip_runtime.h>

// ---------------------------------------------------------------------------
// MI455X (gfx1250) fused MHA: bf16 WMMA GEMMs + flash attention (wave32),
// async global->LDS tile staging (ASYNCcnt) with double-buffered LDS.
// ---------------------------------------------------------------------------

typedef __attribute__((ext_vector_type(16))) __bf16 v16bf;
typedef __attribute__((ext_vector_type(8)))  float  v8f;

#define S_TOK   2048
#define DMODEL  1024
#define NHEAD   16
#define DHEAD   64
#define NROWS   4096   // B * S
#define NEGBIG  (-1.0e9f)

__device__ __forceinline__ unsigned short f32_to_bf16(float f) {
  union { float f; unsigned int u; } c; c.f = f;
  unsigned int r = c.u + 0x7FFFu + ((c.u >> 16) & 1u);   // round-nearest-even
  return (unsigned short)(r >> 16);
}

__device__ __forceinline__ v8f vzero8() { v8f z = {}; return z; }

// A-fragment (16x32 bf16): lane m = L&15, half g = L>>4.
// dword v (0..7) within a 32-wide K window maps to dword index (k = 16g + 2v):
__device__ __forceinline__ int afrag_dword(int g, int v) {
  return g * 8 + v;
}

// Async copy 16B global -> LDS (per lane), tracked by ASYNCcnt.
__device__ __forceinline__ void async_ld_b128(void* ldsdst, const void* src) {
  unsigned loff = (unsigned)(size_t)ldsdst;              // LDS byte offset
  unsigned long long ga = (unsigned long long)(size_t)src;
  asm volatile("global_load_async_to_lds_b128 %0, %1, off"
               :: "v"(loff), "v"(ga) : "memory");
}
__device__ __forceinline__ void wait_async0() {
  asm volatile("s_wait_asynccnt 0x0" ::: "memory");
}

// ---------------------------------------------------------------------------
// fp32 -> bf16 conversion (4 elems / thread, exact grid)
// ---------------------------------------------------------------------------
__global__ void cvt_bf16_kernel(const float* __restrict__ src,
                                unsigned short* __restrict__ dst, int n) {
  int i = (blockIdx.x * blockDim.x + threadIdx.x) * 4;
  if (i + 3 < n) {
    float4 f = *(const float4*)&src[i];
    dst[i + 0] = f32_to_bf16(f.x);
    dst[i + 1] = f32_to_bf16(f.y);
    dst[i + 2] = f32_to_bf16(f.z);
    dst[i + 3] = f32_to_bf16(f.w);
  }
}

// ---------------------------------------------------------------------------
// C[4096x1024] = X[4096x1024](bf16) * W[1024x1024](bf16, KxN row-major) + bias
// mode 0: scatter bf16 into [B,H,S,DK];  mode 1: fp32 row-major out.
// Block: 64x64 tile, 128 threads (4 waves), wave = 32x32 (2x2 WMMA tiles).
// Double-buffered LDS (static indices, k-loop unrolled x2); X tile async.
// ---------------------------------------------------------------------------
__global__ __launch_bounds__(128) void gemm4k_bf16_kernel(
    const unsigned short* __restrict__ X,
    const unsigned short* __restrict__ W,
    const float* __restrict__ bias,
    unsigned short* __restrict__ outBf,
    float* __restrict__ outF,
    int mode) {
  __shared__ __align__(16) unsigned short ldsX[2][64 * 32];   // [m][k]
  __shared__ __align__(16) unsigned short ldsWt[2][64 * 32];  // [n][k]

  const int tid  = threadIdx.x;
  const int lane = tid & 31;
  const int wv   = tid >> 5;
  const int wm   = wv >> 1, wn = wv & 1;
  const int g    = lane >> 4;
  const int l16  = lane & 15;
  const int rowBase = blockIdx.y * 64;
  const int colBase = blockIdx.x * 64;

  v8f acc[2][2];
  acc[0][0] = vzero8(); acc[0][1] = vzero8();
  acc[1][0] = vzero8(); acc[1][1] = vzero8();

  auto loadX = [&](int buf, int k0) {
    #pragma unroll
    for (int i = 0; i < 2; i++) {
      int u = i * 128 + tid;
      int r = (u * 8) >> 5, c = (u * 8) & 31;
      async_ld_b128(&ldsX[buf][r * 32 + c],
                    &X[(rowBase + r) * DMODEL + k0 + c]);
    }
  };
  auto loadW = [&](int buf, int k0) {
    uint4 d[2]; int rr[2], cc[2];
    #pragma unroll
    for (int i = 0; i < 2; i++) {
      int u = i * 128 + tid;
      rr[i] = (u * 8) >> 6; cc[i] = (u * 8) & 63;
      d[i] = *(const uint4*)&W[(k0 + rr[i]) * DMODEL + colBase + cc[i]];
    }
    #pragma unroll
    for (int i = 0; i < 2; i++) {
      const unsigned short* ds = (const unsigned short*)&d[i];
      #pragma unroll
      for (int j = 0; j < 8; j++)
        ldsWt[buf][(cc[i] + j) * 32 + rr[i]] = ds[j];
    }
  };
  auto compute = [&](const unsigned short* bx, const unsigned short* bw) {
    v16bf a[2], b[2];
    #pragma unroll
    for (int mt = 0; mt < 2; mt++) {
      const unsigned int* rp =
          (const unsigned int*)&bx[(wm * 32 + mt * 16 + l16) * 32];
      union { v16bf v; unsigned int u[8]; } fa;
      #pragma unroll
      for (int vv = 0; vv < 8; vv++) fa.u[vv] = rp[afrag_dword(g, vv)];
      a[mt] = fa.v;
    }
    #pragma unroll
    for (int nt = 0; nt < 2; nt++) {
      const uint4* cp =
          (const uint4*)&bw[(wn * 32 + nt * 16 + l16) * 32 + g * 16];
      union { v16bf v; uint4 q[2]; } fb;
      fb.q[0] = cp[0]; fb.q[1] = cp[1];
      b[nt] = fb.v;
    }
    #pragma unroll
    for (int mt = 0; mt < 2; mt++)
      #pragma unroll
      for (int nt = 0; nt < 2; nt++)
        acc[mt][nt] = __builtin_amdgcn_wmma_f32_16x16x32_bf16(
            false, a[mt], false, b[nt], (short)0, acc[mt][nt], false, false);
  };

  loadX(0, 0);
  loadW(0, 0);
  for (int k0 = 0; k0 < DMODEL; k0 += 64) {
    // phase A: compute buf0, prefetch buf1 (k0+32 always < DMODEL here)
    wait_async0();
    __syncthreads();
    __builtin_prefetch(&W[(k0 + 64 + (tid & 31)) * DMODEL + colBase], 0, 3);
    loadX(1, k0 + 32);
    loadW(1, k0 + 32);
    compute(ldsX[0], ldsWt[0]);
    // phase B: compute buf1, prefetch buf0
    wait_async0();
    __syncthreads();
    if (k0 + 64 < DMODEL) {
      loadX(0, k0 + 64);
      loadW(0, k0 + 64);
    }
    compute(ldsX[1], ldsWt[1]);
  }

  // epilogue: +bias, store
  #pragma unroll
  for (int mt = 0; mt < 2; mt++) {
    #pragma unroll
    for (int nt = 0; nt < 2; nt++) {
      int col = colBase + wn * 32 + nt * 16 + l16;
      float bb = bias[col];
      #pragma unroll
      for (int r = 0; r < 8; r++) {
        int row = rowBase + wm * 32 + mt * 16 + g * 8 + r;
        float val = acc[mt][nt][r] + bb;
        if (mode == 0) {
          int bi = row >> 11, s = row & (S_TOK - 1);
          int h = col >> 6, dk = col & (DHEAD - 1);
          outBf[(((size_t)(bi * NHEAD + h) * S_TOK) + s) * DHEAD + dk] =
              f32_to_bf16(val);
        } else {
          outF[(size_t)row * DMODEL + col] = val;
        }
      }
    }
  }
}

// ---------------------------------------------------------------------------
// Flash attention, causal. Block: one (b,h), 64 query rows, 128 threads.
// Wave handles 16 query rows; key blocks of 64 up to the diagonal.
// Double-buffered K/V tiles; K staged via async-to-LDS.
// ---------------------------------------------------------------------------
__global__ __launch_bounds__(128) void attn_kernel(
    const unsigned short* __restrict__ Qp,
    const unsigned short* __restrict__ Kp,
    const unsigned short* __restrict__ Vp,
    unsigned short* __restrict__ ctx) {
  __shared__ __align__(16) unsigned short ldsK[2][64 * 64];     // [key][dk]
  __shared__ __align__(16) unsigned short ldsVt[2][64 * 64];    // [dk][key]
  __shared__ __align__(16) unsigned short ldsP[4 * 16 * 64];    // per-wave P

  const int tid = threadIdx.x, lane = tid & 31, wv = tid >> 5;
  const int g = lane >> 4, l16 = lane & 15;
  const int bh = blockIdx.y, b = bh >> 4, h = bh & 15;
  const int qBase = blockIdx.x * 64;

  const unsigned short* Qh = Qp + (size_t)(b * NHEAD + h) * S_TOK * DHEAD;
  const unsigned short* Kh = Kp + (size_t)(b * NHEAD + h) * S_TOK * DHEAD;
  const unsigned short* Vh = Vp + (size_t)(b * NHEAD + h) * S_TOK * DHEAD;

  auto loadK = [&](int buf, int kBase) {
    #pragma unroll
    for (int i = 0; i < 4; i++) {
      int u = i * 128 + tid;
      int r = (u * 8) >> 6, c = (u * 8) & 63;
      async_ld_b128(&ldsK[buf][r * 64 + c],
                    &Kh[(size_t)(kBase + r) * DHEAD + c]);
    }
  };
  auto loadV = [&](int buf, int kBase) {
    uint4 d[4]; int rr[4], cc[4];
    #pragma unroll
    for (int i = 0; i < 4; i++) {
      int u = i * 128 + tid;
      rr[i] = (u * 8) >> 6; cc[i] = (u * 8) & 63;
      d[i] = *(const uint4*)&Vh[(size_t)(kBase + rr[i]) * DHEAD + cc[i]];
    }
    #pragma unroll
    for (int i = 0; i < 4; i++) {
      const unsigned short* ds = (const unsigned short*)&d[i];
      #pragma unroll
      for (int j = 0; j < 8; j++)
        ldsVt[buf][(cc[i] + j) * 64 + rr[i]] = ds[j];
    }
  };

  // Q A-fragments (2 K-steps over DK=64), loaded straight from global
  v16bf qa[2];
  {
    const int qRow = qBase + wv * 16 + l16;
    const unsigned int* rp = (const unsigned int*)&Qh[(size_t)qRow * DHEAD];
    #pragma unroll
    for (int f = 0; f < 2; f++) {
      union { v16bf v; unsigned int u[8]; } fa;
      #pragma unroll
      for (int vv = 0; vv < 8; vv++) fa.u[vv] = rp[f * 16 + afrag_dword(g, vv)];
      qa[f] = fa.v;
    }
  }

  float mrow[8], lrow[8];
  v8f accO[4];
  #pragma unroll
  for (int i = 0; i < 4; i++) accO[i] = vzero8();
  #pragma unroll
  for (int r = 0; r < 8; r++) { mrow[r] = -3.0e38f; lrow[r] = 0.f; }

  const int diag = qBase >> 6;
  loadK(0, 0);
  loadV(0, 0);
  int cur = 0;
  for (int kb = 0; kb <= diag; kb++) {
    const int kBase = kb * 64;
    wait_async0();
    __syncthreads();
    if (kb < diag) {
      loadK(cur ^ 1, kBase + 64);
      loadV(cur ^ 1, kBase + 64);
    }

    // scores S = Q * K^T  (16 rows x 64 keys per wave)
    v8f sc[4];
    #pragma unroll
    for (int nt = 0; nt < 4; nt++) {
      v8f s = vzero8();
      #pragma unroll
      for (int f = 0; f < 2; f++) {
        const uint4* cp =
            (const uint4*)&ldsK[cur][(nt * 16 + l16) * 64 + f * 32 + g * 16];
        union { v16bf v; uint4 q[2]; } fb;
        fb.q[0] = cp[0]; fb.q[1] = cp[1];
        s = __builtin_amdgcn_wmma_f32_16x16x32_bf16(
            false, qa[f], false, fb.v, (short)0, s, false, false);
      }
      sc[nt] = s;
    }

    // scale 1/sqrt(64) and causal mask on diagonal block
    const bool diagBlk = (kb == diag);
    #pragma unroll
    for (int nt = 0; nt < 4; nt++) {
      int col = kBase + nt * 16 + l16;
      #pragma unroll
      for (int r = 0; r < 8; r++) {
        float s = sc[nt][r] * 0.125f;
        if (diagBlk) {
          int row = qBase + wv * 16 + g * 8 + r;
          if (col > row) s = NEGBIG;
        }
        sc[nt][r] = s;
      }
    }

    // online softmax per row (rows replicated across the 16-lane half)
    #pragma unroll
    for (int r = 0; r < 8; r++) {
      float mx = fmaxf(fmaxf(sc[0][r], sc[1][r]), fmaxf(sc[2][r], sc[3][r]));
      #pragma unroll
      for (int msk = 1; msk < 16; msk <<= 1)
        mx = fmaxf(mx, __shfl_xor(mx, msk, 32));
      float mnew = fmaxf(mrow[r], mx);
      float corr = __expf(mrow[r] - mnew);
      float psum = 0.f;
      #pragma unroll
      for (int nt = 0; nt < 4; nt++) {
        float p = __expf(sc[nt][r] - mnew);
        sc[nt][r] = p;
        psum += p;
      }
      #pragma unroll
      for (int msk = 1; msk < 16; msk <<= 1)
        psum += __shfl_xor(psum, msk, 32);
      lrow[r] = lrow[r] * corr + psum;
      mrow[r] = mnew;
      #pragma unroll
      for (int d = 0; d < 4; d++) accO[d][r] *= corr;
    }

    // stage P (C-layout -> LDS row-major) in per-wave region (wave-private)
    unsigned short* Pw = ldsP + wv * (16 * 64);
    #pragma unroll
    for (int nt = 0; nt < 4; nt++)
      #pragma unroll
      for (int r = 0; r < 8; r++)
        Pw[(g * 8 + r) * 64 + nt * 16 + l16] = f32_to_bf16(sc[nt][r]);

    // O += P * V
    #pragma unroll
    for (int f = 0; f < 2; f++) {
      const unsigned int* rp = (const unsigned int*)&Pw[l16 * 64];
      union { v16bf v; unsigned int u[8]; } fa;
      #pragma unroll
      for (int vv = 0; vv < 8; vv++) fa.u[vv] = rp[f * 16 + afrag_dword(g, vv)];
      v16bf pa = fa.v;
      #pragma unroll
      for (int nt = 0; nt < 4; nt++) {
        const uint4* cp =
            (const uint4*)&ldsVt[cur][(nt * 16 + l16) * 64 + f * 32 + g * 16];
        union { v16bf v; uint4 q[2]; } fb;
        fb.q[0] = cp[0]; fb.q[1] = cp[1];
        accO[nt] = __builtin_amdgcn_wmma_f32_16x16x32_bf16(
            false, pa, false, fb.v, (short)0, accO[nt], false, false);
      }
    }
    cur ^= 1;
  }

  // epilogue: normalize, store ctx bf16 [B,S,D]
  #pragma unroll
  for (int nt = 0; nt < 4; nt++) {
    #pragma unroll
    for (int r = 0; r < 8; r++) {
      int srow = qBase + wv * 16 + g * 8 + r;
      float val = accO[nt][r] / lrow[r];
      ctx[((size_t)(b * S_TOK + srow)) * DMODEL + h * DHEAD + nt * 16 + l16] =
          f32_to_bf16(val);
    }
  }
}

// ---------------------------------------------------------------------------
// Host launcher
// ---------------------------------------------------------------------------
extern "C" void kernel_launch(void* const* d_in, const int* in_sizes, int n_in,
                              void* d_out, int out_size, void* d_ws,
                              size_t ws_size, hipStream_t stream) {
  const float* q   = (const float*)d_in[0];
  const float* k   = (const float*)d_in[1];
  const float* v   = (const float*)d_in[2];
  // d_in[3] = mask (causal known statically, unused)
  const float* w_q = (const float*)d_in[4];
  const float* b_q = (const float*)d_in[5];
  const float* w_k = (const float*)d_in[6];
  const float* b_k = (const float*)d_in[7];
  const float* w_v = (const float*)d_in[8];
  const float* b_v = (const float*)d_in[9];
  const float* w_o = (const float*)d_in[10];
  const float* b_o = (const float*)d_in[11];
  float* out = (float*)d_out;

  const size_t ACT = (size_t)NROWS * DMODEL;   // 4M elems
  const size_t WSZ = (size_t)DMODEL * DMODEL;  // 1M elems
  unsigned short* ws  = (unsigned short*)d_ws;
  unsigned short* qb  = ws;
  unsigned short* kb  = qb + ACT;
  unsigned short* vb  = kb + ACT;
  unsigned short* wqb = vb + ACT;
  unsigned short* wkb = wqb + WSZ;
  unsigned short* wvb = wkb + WSZ;
  unsigned short* wob = wvb + WSZ;
  unsigned short* Qp  = wob + WSZ;
  unsigned short* Kp  = Qp + ACT;
  unsigned short* Vp  = Kp + ACT;
  unsigned short* ctx = qb;  // reuse q-bf16 region after projections

  // conversions (fp32 -> bf16)
  cvt_bf16_kernel<<<(int)(ACT / 1024), 256, 0, stream>>>(q, qb, (int)ACT);
  cvt_bf16_kernel<<<(int)(ACT / 1024), 256, 0, stream>>>(k, kb, (int)ACT);
  cvt_bf16_kernel<<<(int)(ACT / 1024), 256, 0, stream>>>(v, vb, (int)ACT);
  cvt_bf16_kernel<<<(int)(WSZ / 1024), 256, 0, stream>>>(w_q, wqb, (int)WSZ);
  cvt_bf16_kernel<<<(int)(WSZ / 1024), 256, 0, stream>>>(w_k, wkb, (int)WSZ);
  cvt_bf16_kernel<<<(int)(WSZ / 1024), 256, 0, stream>>>(w_v, wvb, (int)WSZ);
  cvt_bf16_kernel<<<(int)(WSZ / 1024), 256, 0, stream>>>(w_o, wob, (int)WSZ);

  // QKV projections -> [B,H,S,DK] bf16
  dim3 gg(DMODEL / 64, NROWS / 64), gb(128);
  gemm4k_bf16_kernel<<<gg, gb, 0, stream>>>(qb, wqb, b_q, Qp, nullptr, 0);
  gemm4k_bf16_kernel<<<gg, gb, 0, stream>>>(kb, wkb, b_k, Kp, nullptr, 0);
  gemm4k_bf16_kernel<<<gg, gb, 0, stream>>>(vb, wvb, b_v, Vp, nullptr, 0);

  // flash attention -> ctx bf16 [B,S,D]
  attn_kernel<<<dim3(S_TOK / 64, 2 * NHEAD), 128, 0, stream>>>(Qp, Kp, Vp, ctx);

  // output projection -> fp32 out
  gemm4k_bf16_kernel<<<gg, gb, 0, stream>>>(ctx, wob, b_o, nullptr, out, 1);
}
